// MultiheadAttention_16939351015915
// MI455X (gfx1250) — compile-verified, matrix operand order checked by reading, not run on hardware
//
#include <hip/hip_runtime.h>
#include <hip/hip_bf16.h>

// ---------------------------------------------------------------------------
// Problem constants (from reference)
// ---------------------------------------------------------------------------
constexpr int Bc   = 4;
constexpr int Sc   = 1024;
constexpr int Dc   = 768;
constexpr int NHc  = 12;
constexpr int DHc  = 64;
constexpr int CLSc = 8;
constexpr int MAXPc = 1024;
constexpr int Rc   = 2048;          // 2*MAXP
constexpr int HPDc = 384;           // (NH/2)*DH  (positional projection width)
constexpr float SCALEc = 0.05590169943749474f;   // 1/sqrt(64*5)
constexpr float POS_SCALEc = 0.05590169943749474f;

typedef __attribute__((ext_vector_type(16))) __bf16 v16bf;
typedef __attribute__((ext_vector_type(8)))  float  v8f;

union FragU { unsigned u[8]; v16bf v; };

#if defined(__has_builtin)
#if __has_builtin(__builtin_amdgcn_global_load_async_to_lds_b128)
#define HAVE_ASYNC_LDS 1
#endif
#endif

// builtin expects: (int4 addrspace(1)*, int4 addrspace(3)*, imm offset, imm cpol)
typedef int v4i __attribute__((vector_size(16)));
typedef __attribute__((address_space(1))) v4i* as1_v4ip;
typedef __attribute__((address_space(3))) v4i* as3_v4ip;

// Stage 16 bytes global -> LDS. Async (GLOBAL_LOAD_ASYNC_TO_LDS_B128, ASYNCcnt)
// when the builtin exists, else a plain synchronous copy.
__device__ __forceinline__ void stage16(const __bf16* g, __bf16* l) {
#ifdef HAVE_ASYNC_LDS
  __builtin_amdgcn_global_load_async_to_lds_b128(
      (as1_v4ip)(v4i*)(void*)const_cast<__bf16*>(g),
      (as3_v4ip)(v4i*)(void*)l, 0, 0);
#else
  *reinterpret_cast<uint4*>(l) = *reinterpret_cast<const uint4*>(g);
#endif
}

__device__ __forceinline__ void stage_flush() {
#ifdef HAVE_ASYNC_LDS
#if __has_builtin(__builtin_amdgcn_s_wait_asynccnt)
  __builtin_amdgcn_s_wait_asynccnt(0);
#else
  asm volatile("s_wait_asynccnt 0" ::: "memory");
#endif
#endif
}

// K index pattern for 16-bit A/B fragments (ISA 7.12.2, 16-bit A 16x32):
// VGPR v in [0..3]: K = 8*half + 2v ; v in [4..7]: K = 16 + 8*half + 2(v-4)
__device__ __forceinline__ int frag_k(int v, int hl) {
  return (v < 4) ? (8 * hl + 2 * v) : (16 + 8 * hl + 2 * (v - 4));
}

// Row-major fragment load from global: two 16B runs -> 2x global_load_b128
__device__ __forceinline__ v16bf load_frag_row(const __bf16* __restrict__ p,
                                               long row, int k0, int ld, int lane) {
  int hl = (lane >> 4) & 1;
  const __bf16* base = p + row * (long)ld + k0;
  FragU f;
#pragma unroll
  for (int v = 0; v < 8; ++v)
    f.u[v] = *reinterpret_cast<const unsigned*>(base + frag_k(v, hl));
  return f.v;
}

__device__ __forceinline__ v8f wmma_bf16(v16bf a, v16bf b, v8f c) {
  return __builtin_amdgcn_wmma_f32_16x16x32_bf16(false, a, false, b, (short)0, c,
                                                 false, false);
}

// ---------------------------------------------------------------------------
// fp32 -> bf16 conversion
// ---------------------------------------------------------------------------
__global__ void cvt_bf16_kernel(const float* __restrict__ src,
                                __bf16* __restrict__ dst, int n) {
  int i = blockIdx.x * blockDim.x + threadIdx.x;
  if (i < n) dst[i] = (__bf16)src[i];
}

// bq2[n] = bq[n] + r_w_bias.flat[n]
__global__ void bias_combine_kernel(const float* __restrict__ bq,
                                    const float* __restrict__ rwb,
                                    float* __restrict__ bq2, int n) {
  int i = blockIdx.x * blockDim.x + threadIdx.x;
  if (i < n) bq2[i] = bq[i] + rwb[i];
}

// ---------------------------------------------------------------------------
// LayerNorm (one wave per row, 768 cols) -> bf16
// ---------------------------------------------------------------------------
__global__ __launch_bounds__(128) void ln_bf16_kernel(
    const float* __restrict__ x, const float* __restrict__ gamma,
    const float* __restrict__ beta, __bf16* __restrict__ y, int rows, int cols) {
  int lane = threadIdx.x & 31;
  int wid  = threadIdx.x >> 5;
  int row  = blockIdx.x * 4 + wid;
  if (row >= rows) return;
  const float* xr = x + (long)row * cols;
  float s = 0.f, s2 = 0.f;
  for (int c = lane; c < cols; c += 32) { float v = xr[c]; s += v; s2 += v * v; }
#pragma unroll
  for (int o = 16; o >= 1; o >>= 1) {
    s  += __shfl_xor(s, o, 32);
    s2 += __shfl_xor(s2, o, 32);
  }
  float mean = s / cols;
  float var  = s2 / cols - mean * mean;
  float rstd = rsqrtf(var + 1e-12f);
  __bf16* yr = y + (long)row * cols;
  for (int c = lane; c < cols; c += 32)
    yr[c] = (__bf16)((xr[c] - mean) * rstd * gamma[c] + beta[c]);
}

// ---------------------------------------------------------------------------
// Generic bf16 WMMA GEMM:  Dst = (A[M,K] * W[N,K]^T + bias[N]) * alpha
// One wave computes a 32x32 tile.
// storeT == 0 : Dst[M,N] row-major bf16
// storeT == 1 : Dst is [B][NH][DH][S] (head-transposed V layout)
// ---------------------------------------------------------------------------
__global__ __launch_bounds__(128) void gemm_bf16_kernel(
    const __bf16* __restrict__ A, const __bf16* __restrict__ W,
    const float* __restrict__ bias, __bf16* __restrict__ Dst,
    int M, int N, int K, float alpha, int storeT) {
  int lane = threadIdx.x & 31;
  int wid  = threadIdx.x >> 5;
  int tile = blockIdx.x * 4 + wid;
  int tn_cnt = N >> 5;
  int tmr = (tile / tn_cnt) << 5;
  int tnr = (tile % tn_cnt) << 5;
  if (tmr >= M) return;
  int r = lane & 15;
  v8f acc00 = {}, acc01 = {}, acc10 = {}, acc11 = {};
  for (int k0 = 0; k0 < K; k0 += 32) {
    v16bf a0 = load_frag_row(A, tmr + r,      k0, K, lane);
    v16bf a1 = load_frag_row(A, tmr + 16 + r, k0, K, lane);
    v16bf b0 = load_frag_row(W, tnr + r,      k0, K, lane);
    v16bf b1 = load_frag_row(W, tnr + 16 + r, k0, K, lane);
    acc00 = wmma_bf16(a0, b0, acc00);
    acc01 = wmma_bf16(a0, b1, acc01);
    acc10 = wmma_bf16(a1, b0, acc10);
    acc11 = wmma_bf16(a1, b1, acc11);
  }
  int hl = lane >> 4;
  float bv0 = bias ? bias[tnr + r]      : 0.f;
  float bv1 = bias ? bias[tnr + 16 + r] : 0.f;
  if (storeT == 0) {
#pragma unroll
    for (int g = 0; g < 8; ++g) {
      long m0 = tmr + g + 8 * hl;
      long m1 = m0 + 16;
      Dst[m0 * N + tnr + r]      = (__bf16)((acc00[g] + bv0) * alpha);
      Dst[m0 * N + tnr + 16 + r] = (__bf16)((acc01[g] + bv1) * alpha);
      Dst[m1 * N + tnr + r]      = (__bf16)((acc10[g] + bv0) * alpha);
      Dst[m1 * N + tnr + 16 + r] = (__bf16)((acc11[g] + bv1) * alpha);
    }
  } else {
    // head-transposed store: (mrow=(b*S+s), ncol=(n*DH+dh)) -> [b][n][dh][s]
#pragma unroll
    for (int g = 0; g < 8; ++g) {
      long m0 = tmr + g + 8 * hl;
      long m1 = m0 + 16;
      long n0 = tnr + r, n1 = tnr + 16 + r;
      long b0i = m0 >> 10, s0i = m0 & (Sc - 1);
      long b1i = m1 >> 10, s1i = m1 & (Sc - 1);
      Dst[((b0i * NHc + (n0 >> 6)) * DHc + (n0 & 63)) * Sc + s0i] =
          (__bf16)((acc00[g] + bv0) * alpha);
      Dst[((b0i * NHc + (n1 >> 6)) * DHc + (n1 & 63)) * Sc + s0i] =
          (__bf16)((acc01[g] + bv1) * alpha);
      Dst[((b1i * NHc + (n0 >> 6)) * DHc + (n0 & 63)) * Sc + s1i] =
          (__bf16)((acc10[g] + bv0) * alpha);
      Dst[((b1i * NHc + (n1 >> 6)) * DHc + (n1 & 63)) * Sc + s1i] =
          (__bf16)((acc11[g] + bv1) * alpha);
    }
  }
}

// ---------------------------------------------------------------------------
// Flash attention with disentangled 2D positional bias.
// Block = 4 waves sharing (b, head), adjacent 16-row q-tiles. K/V tiles are
// staged cooperatively into LDS via async global->LDS copies (ASYNCcnt), so
// each tile is fetched once per block instead of once per wave.
// ---------------------------------------------------------------------------
__global__ __launch_bounds__(128) void attn_kernel(
    const __bf16* __restrict__ qh, const __bf16* __restrict__ kh,
    const __bf16* __restrict__ vT, const __bf16* __restrict__ EKrow,
    const __bf16* __restrict__ EQrow, const __bf16* __restrict__ Ecol,
    const float* __restrict__ query, float* __restrict__ out) {
  int lane = threadIdx.x & 31;
  int wid  = threadIdx.x >> 5;
  int hl   = lane >> 4;
  int r15  = lane & 15;

  int gw = blockIdx.x * 4 + wid;            // 0..3071
  int qt = gw & 63;                         // S/16 = 64 q-tiles
  int n  = (gw >> 6) % NHc;
  int b  = gw / (64 * NHc);
  int i0 = qt << 4;

  const __bf16* EK; const __bf16* EQ; int h;
  if (n < NHc / 2) { EK = EKrow; EQ = EQrow; h = n; }
  else             { EK = Ecol;  EQ = Ecol;  h = n - NHc / 2; }
  const __bf16* EKh = EK + h * DHc;
  const __bf16* EQh = EQ + h * DHc;

  // per-wave scratch for the diagonal gather and P-layout transpose
  __shared__ alignas(16) float  ldsT1[4][16][33];
  __shared__ alignas(16) float  ldsT2[4][16][33];
  __shared__ alignas(16) __bf16 ldsP[4][16][36];
  // block-shared staged tiles (row strides keep 16B-aligned async writes)
  __shared__ alignas(16) __bf16 ldsK[32][72];   // [key_local][feature]
  __shared__ alignas(16) __bf16 ldsV[64][40];   // [dh][key_local]

  const __bf16* qbase  = qh + (long)b * Sc * Dc + n * DHc;
  const __bf16* kbase  = kh + (long)b * Sc * Dc + n * DHc;
  const __bf16* vtbase = vT + ((long)b * NHc + n) * DHc * Sc;

  v16bf Qf0 = load_frag_row(qbase, i0 + r15, 0,  Dc, lane);
  v16bf Qf1 = load_frag_row(qbase, i0 + r15, 32, Dc, lane);

  v8f O0 = {}, O1 = {}, O2 = {}, O3 = {};
  float mstat[8], lstat[8];
#pragma unroll
  for (int g = 0; g < 8; ++g) { mstat[g] = -1e30f; lstat[g] = 0.f; }

  for (int jt = 0; jt < Sc; jt += 32) {
    // ---- cooperative async staging: K tile (32x64), V^T tile (64x32) ----
    {
      int t = threadIdx.x;
#pragma unroll
      for (int i = 0; i < 2; ++i) {
        int c  = t + i * 128;
        int kr = c >> 3, kc = (c & 7) * 8;       // 32 rows x 8 chunks
        stage16(kbase + (long)(jt + kr) * Dc + kc, &ldsK[kr][kc]);
        int vr = c >> 2, vc = (c & 3) * 8;       // 64 rows x 4 chunks
        stage16(vtbase + (long)vr * Sc + jt + vc, &ldsV[vr][vc]);
      }
      stage_flush();
      __syncthreads();
    }

    float sc[2][8];
#pragma unroll
    for (int st = 0; st < 2; ++st) {
      int j0 = jt + 16 * st;
      // K fragments from LDS (shared across the 4 waves)
      FragU kf0, kf1;
#pragma unroll
      for (int v = 0; v < 8; ++v) {
        kf0.u[v] = *reinterpret_cast<const unsigned*>(
            &ldsK[st * 16 + r15][frag_k(v, hl)]);
        kf1.u[v] = *reinterpret_cast<const unsigned*>(
            &ldsK[st * 16 + r15][32 + frag_k(v, hl)]);
      }
      v16bf Kf0 = kf0.v, Kf1 = kf1.v;
      // content scores (qh already carries the global scale)
      v8f s = {};
      s = wmma_bf16(Qf0, Kf0, s);
      s = wmma_bf16(Qf1, Kf1, s);
      // positional band: T1[i,c] = q[i].EK[r0+c], T2[j,c] = k[j].EQ[s0+c]
      int d0  = i0 - j0;
      int rr0 = d0 + MAXPc - 15;
      int ss0 = -d0 + MAXPc - 15;
      v8f t1a = {}, t1b = {}, t2a = {}, t2b = {};
      v16bf e;
      e = load_frag_row(EKh, rr0 + r15,      0,  HPDc, lane); t1a = wmma_bf16(Qf0, e, t1a);
      e = load_frag_row(EKh, rr0 + r15,      32, HPDc, lane); t1a = wmma_bf16(Qf1, e, t1a);
      e = load_frag_row(EKh, rr0 + 16 + r15, 0,  HPDc, lane); t1b = wmma_bf16(Qf0, e, t1b);
      e = load_frag_row(EKh, rr0 + 16 + r15, 32, HPDc, lane); t1b = wmma_bf16(Qf1, e, t1b);
      e = load_frag_row(EQh, ss0 + r15,      0,  HPDc, lane); t2a = wmma_bf16(Kf0, e, t2a);
      e = load_frag_row(EQh, ss0 + r15,      32, HPDc, lane); t2a = wmma_bf16(Kf1, e, t2a);
      e = load_frag_row(EQh, ss0 + 16 + r15, 0,  HPDc, lane); t2b = wmma_bf16(Kf0, e, t2b);
      e = load_frag_row(EQh, ss0 + 16 + r15, 32, HPDc, lane); t2b = wmma_bf16(Kf1, e, t2b);
      // spill band tiles to LDS for the diagonal gather
#pragma unroll
      for (int g = 0; g < 8; ++g) {
        int m = g + 8 * hl;
        ldsT1[wid][m][r15]      = t1a[g];
        ldsT1[wid][m][16 + r15] = t1b[g];
        ldsT2[wid][m][r15]      = t2a[g];
        ldsT2[wid][m][16 + r15] = t2b[g];
      }
      asm volatile("s_wait_dscnt 0" ::: "memory");
#pragma unroll
      for (int g = 0; g < 8; ++g) {
        int m = g + 8 * hl;
        int i = i0 + m, j = j0 + r15;
        float bias = 0.f;
        if (i >= CLSc && j >= CLSc)
          bias = ldsT1[wid][m][m - r15 + 15] +
                 POS_SCALEc * ldsT2[wid][r15][r15 - m + 15];
        sc[st][g] = s[g] + bias;
      }
      asm volatile("s_wait_dscnt 0" ::: "memory");
    }
    // online softmax over the 32 fresh keys
#pragma unroll
    for (int g = 0; g < 8; ++g) {
      float x = fmaxf(sc[0][g], sc[1][g]);
#pragma unroll
      for (int o = 8; o >= 1; o >>= 1) x = fmaxf(x, __shfl_xor(x, o, 32));
      float mold = mstat[g];
      float mnew = fmaxf(mold, x);
      float corr = __expf(mold - mnew);
      mstat[g] = mnew;
      float p0 = __expf(sc[0][g] - mnew);
      float p1 = __expf(sc[1][g] - mnew);
      float rs = p0 + p1;
#pragma unroll
      for (int o = 8; o >= 1; o >>= 1) rs += __shfl_xor(rs, o, 32);
      lstat[g] = lstat[g] * corr + rs;
      O0[g] *= corr; O1[g] *= corr; O2[g] *= corr; O3[g] *= corr;
      int m = g + 8 * hl;
      ldsP[wid][m][r15]      = (__bf16)p0;
      ldsP[wid][m][16 + r15] = (__bf16)p1;
    }
    asm volatile("s_wait_dscnt 0" ::: "memory");
    FragU pf;
#pragma unroll
    for (int v = 0; v < 8; ++v)
      pf.u[v] = *reinterpret_cast<const unsigned*>(&ldsP[wid][r15][frag_k(v, hl)]);
    // P(16x32) x V(32x64): V fragments from the staged LDS tile
#pragma unroll
    for (int t4 = 0; t4 < 4; ++t4) {
      FragU vf;
#pragma unroll
      for (int v = 0; v < 8; ++v)
        vf.u[v] = *reinterpret_cast<const unsigned*>(
            &ldsV[t4 * 16 + r15][frag_k(v, hl)]);
      if (t4 == 0) O0 = wmma_bf16(pf.v, vf.v, O0);
      else if (t4 == 1) O1 = wmma_bf16(pf.v, vf.v, O1);
      else if (t4 == 2) O2 = wmma_bf16(pf.v, vf.v, O2);
      else O3 = wmma_bf16(pf.v, vf.v, O3);
    }
    if (jt + 32 < Sc)
      __builtin_prefetch(kbase + (long)(jt + 32 + r15) * Dc, 0, 0);
    __syncthreads();   // staged tiles fully consumed before next overwrite
  }
  // epilogue: normalize, residual add
#pragma unroll
  for (int g = 0; g < 8; ++g) {
    float inv = 1.f / lstat[g];
    int m = g + 8 * hl;
    long base = ((long)b * Sc + i0 + m) * Dc + n * DHc + r15;
    out[base]      = query[base]      + O0[g] * inv;
    out[base + 16] = query[base + 16] + O1[g] * inv;
    out[base + 32] = query[base + 32] + O2[g] * inv;
    out[base + 48] = query[base + 48] + O3[g] * inv;
  }
}

// ---------------------------------------------------------------------------
// Host-side launcher
// ---------------------------------------------------------------------------
extern "C" void kernel_launch(void* const* d_in, const int* in_sizes, int n_in,
                              void* d_out, int out_size, void* d_ws, size_t ws_size,
                              hipStream_t stream) {
  (void)in_sizes; (void)n_in; (void)out_size; (void)ws_size;
  const float* query = (const float*)d_in[0];
  const float* key_  = (const float*)d_in[1];
  const float* value = (const float*)d_in[2];
  const float* rowE  = (const float*)d_in[3];
  const float* colE  = (const float*)d_in[4];
  const float* Wq    = (const float*)d_in[5];
  const float* bq    = (const float*)d_in[6];
  const float* Wk    = (const float*)d_in[7];
  const float* Wv    = (const float*)d_in[8];
  const float* bv    = (const float*)d_in[9];
  const float* Wpq   = (const float*)d_in[10];
  const float* bpq   = (const float*)d_in[11];
  const float* Wpk   = (const float*)d_in[12];
  const float* rwb   = (const float*)d_in[13];
  const float* ln_g  = (const float*)d_in[14];
  const float* ln_b  = (const float*)d_in[15];
  float* out = (float*)d_out;

  const int BS = Bc * Sc;                       // 4096
  char* ws = (char*)d_ws;
  size_t off = 0;
  auto carve = [&](size_t bytes) -> void* {
    off = (off + 255) & ~(size_t)255;
    void* p = ws + off;
    off += bytes;
    return p;
  };

  __bf16* q_ln  = (__bf16*)carve((size_t)BS * Dc * 2);
  __bf16* k_ln  = (__bf16*)carve((size_t)BS * Dc * 2);
  __bf16* v_ln  = (__bf16*)carve((size_t)BS * Dc * 2);
  __bf16* Wq_b  = (__bf16*)carve((size_t)Dc * Dc * 2);
  __bf16* Wk_b  = (__bf16*)carve((size_t)Dc * Dc * 2);
  __bf16* Wv_b  = (__bf16*)carve((size_t)Dc * Dc * 2);
  __bf16* Wpq_b = (__bf16*)carve((size_t)HPDc * Dc * 2);
  __bf16* Wpk_b = (__bf16*)carve((size_t)HPDc * Dc * 2);
  __bf16* rowEb = (__bf16*)carve((size_t)Rc * Dc * 2);
  __bf16* colEb = (__bf16*)carve((size_t)Rc * Dc * 2);
  __bf16* qhB   = (__bf16*)carve((size_t)BS * Dc * 2);
  __bf16* khB   = (__bf16*)carve((size_t)BS * Dc * 2);
  __bf16* vTB   = (__bf16*)carve((size_t)BS * Dc * 2);          // [B][NH][DH][S]
  __bf16* EKrow = (__bf16*)carve((size_t)(Rc + 1) * HPDc * 2);  // +1 pad row for band
  __bf16* EQrow = (__bf16*)carve((size_t)(Rc + 1) * HPDc * 2);
  __bf16* Ecol  = (__bf16*)carve((size_t)(Rc + 1) * HPDc * 2);
  float*  bq2   = (float*)carve((size_t)Dc * 4);

  auto cvt = [&](const float* s, __bf16* d, int n) {
    cvt_bf16_kernel<<<(n + 255) / 256, 256, 0, stream>>>(s, d, n);
  };
  cvt(Wq,  Wq_b,  Dc * Dc);
  cvt(Wk,  Wk_b,  Dc * Dc);
  cvt(Wv,  Wv_b,  Dc * Dc);
  cvt(Wpq, Wpq_b, HPDc * Dc);
  cvt(Wpk, Wpk_b, HPDc * Dc);
  cvt(rowE, rowEb, Rc * Dc);
  cvt(colE, colEb, Rc * Dc);
  bias_combine_kernel<<<(Dc + 255) / 256, 256, 0, stream>>>(bq, rwb, bq2, Dc);

  ln_bf16_kernel<<<BS / 4, 128, 0, stream>>>(query, ln_g, ln_b, q_ln, BS, Dc);
  ln_bf16_kernel<<<BS / 4, 128, 0, stream>>>(key_,  ln_g, ln_b, k_ln, BS, Dc);
  ln_bf16_kernel<<<BS / 4, 128, 0, stream>>>(value, ln_g, ln_b, v_ln, BS, Dc);

  auto gemm = [&](const __bf16* A, const __bf16* W, const float* bias,
                  __bf16* Dst, int M, int N, int K, float alpha, int storeT) {
    int tiles = (M / 32) * (N / 32);
    gemm_bf16_kernel<<<(tiles + 3) / 4, 128, 0, stream>>>(A, W, bias, Dst, M, N, K,
                                                          alpha, storeT);
  };
  // qh = (q_ln*Wq^T + bq + r_w_bias) * scale   (scaled q, used by content AND c2p)
  gemm(q_ln, Wq_b, bq2, qhB, BS, Dc, Dc, SCALEc, 0);
  // kh = k_ln*Wk^T
  gemm(k_ln, Wk_b, nullptr, khB, BS, Dc, Dc, 1.f, 0);
  // vT = head-transposed (v_ln*Wv^T + bv)
  gemm(v_ln, Wv_b, bv, vTB, BS, Dc, Dc, 1.f, 1);
  // positional projections (col_qp == col_k, so only 3 needed)
  gemm(rowEb, Wpk_b, nullptr, EKrow, Rc, HPDc, Dc, 1.f, 0);   // row_k
  gemm(rowEb, Wpq_b, bpq,     EQrow, Rc, HPDc, Dc, 1.f, 0);   // row_qp
  gemm(colEb, Wpq_b, bpq,     Ecol,  Rc, HPDc, Dc, 1.f, 0);   // col_k == col_qp

  // attention: B*NH*(S/16) = 3072 waves, 4 waves per block
  attn_kernel<<<(Bc * NHc * (Sc / 16)) / 4, 128, 0, stream>>>(
      qhB, khB, vTB, EKrow, EQrow, Ecol, query, out);
}
